// TemporalAdapter_30485677868020
// MI455X (gfx1250) — matrix operand order repaired and checked by reading, not compiled, not run on hardware
//
#include <hip/hip_runtime.h>
#include <hip/hip_bf16.h>
#include <math.h>

#define B_  8
#define T_  2048
#define D_  2048
#define K_  128
#define M_  8
#define DC_ 256
#define NTOK (B_ * T_)

#if __has_builtin(__builtin_amdgcn_global_load_async_to_lds_b128)
#define HAS_ASYNC 1
#else
#define HAS_ASYNC 0
#endif

typedef __attribute__((ext_vector_type(16))) __bf16 v16bf;
typedef __attribute__((ext_vector_type(8)))  float  v8f;
typedef __attribute__((ext_vector_type(4)))  unsigned int v4u;
typedef __attribute__((ext_vector_type(4)))  int v4i;

__device__ __forceinline__ unsigned short f2bf(float f) {
    unsigned int x = __float_as_uint(f);
    unsigned int r = x + 0x7FFFu + ((x >> 16) & 1u);   // RNE
    return (unsigned short)(r >> 16);
}
__device__ __forceinline__ float bf2f(unsigned short h) {
    return __uint_as_float(((unsigned int)h) << 16);
}

#if HAS_ASYNC
#define AS3 __attribute__((address_space(3)))
__device__ __forceinline__ void async_copy16(const void* g, void* lds) {
    // generic LDS pointer: low 32 bits are the LDS byte offset (aperture form)
    __builtin_amdgcn_global_load_async_to_lds_b128(
        (v4i*)(unsigned long long)g,
        (AS3 v4i*)(unsigned int)(unsigned long long)lds,
        0, 0);
}
#endif

// ------------------------------------------- fp32 [K][P] -> bf16 [P][K] (K-major)
__global__ __launch_bounds__(256)
void transpose_bf16_kernel(const float* __restrict__ in,
                           unsigned short* __restrict__ out,
                           int Krows, int Pcols) {
    __shared__ unsigned short tile[32][33];
    const int p0 = blockIdx.x * 32, k0 = blockIdx.y * 32;
    const int tx = threadIdx.x & 31, ty = threadIdx.x >> 5;   // 32 x 8
#pragma unroll
    for (int j = 0; j < 32; j += 8)
        tile[ty + j][tx] = f2bf(in[(size_t)(k0 + ty + j) * Pcols + p0 + tx]);
    __syncthreads();
#pragma unroll
    for (int j = 0; j < 32; j += 8)
        out[(size_t)(p0 + ty + j) * Krows + k0 + tx] = tile[tx][ty + j];
}

// ------------------------------------------------------- tun (bf16) features
__global__ void tun_kernel(const float* __restrict__ delta,
                           const float* __restrict__ mu,
                           const float* __restrict__ log_sigma,
                           unsigned short* __restrict__ tun_b) {
    int idx = blockIdx.x * 256 + threadIdx.x;        // over NTOK*K_
    int n = idx >> 7, k = idx & (K_ - 1);
    float ld = log1pf(fmaxf(delta[n], 0.0f));
    float z  = (ld - mu[k]) / (expf(log_sigma[k]) + 1e-6f);
    tun_b[idx] = f2bf(expf(-0.5f * z * z));
}

// ------------------------------------------------------------- alpha factors
__global__ void alpha_kernel(const float* __restrict__ delta,
                             float* __restrict__ alpha) {
    int idx = blockIdx.x * 256 + threadIdx.x;        // over NTOK*M_
    int m = idx & (M_ - 1);
    int n = idx >> 3;
    int t = n & (T_ - 1);
    float dt = (t > 0) ? fmaxf(delta[n] - delta[n - 1], 0.0f) : 0.0f;
    float tau = expf((float)m * (logf(86400.0f) / (float)(M_ - 1)));
    alpha[idx] = expf(-dt / tau);
}

// ------------------------------------------- LayerNorm; out = h + h_norm seed
__global__ __launch_bounds__(256)
void ln_kernel(const float* __restrict__ h, const float* __restrict__ gamma,
               const float* __restrict__ beta,
               unsigned short* __restrict__ hnorm_b, float* __restrict__ out) {
    __shared__ float red[256], red2[256];
    const int n = blockIdx.x;
    const float* hr = h + (size_t)n * D_;
    float vals[8], s = 0.f, s2 = 0.f;
#pragma unroll
    for (int j = 0; j < 8; ++j) {
        float x = hr[threadIdx.x + j * 256];
        vals[j] = x; s += x; s2 += x * x;
    }
    red[threadIdx.x] = s; red2[threadIdx.x] = s2;
    __syncthreads();
    for (int off = 128; off > 0; off >>= 1) {
        if (threadIdx.x < off) {
            red[threadIdx.x]  += red[threadIdx.x + off];
            red2[threadIdx.x] += red2[threadIdx.x + off];
        }
        __syncthreads();
    }
    float mean = red[0] * (1.0f / D_);
    float var  = red2[0] * (1.0f / D_) - mean * mean;
    float rstd = rsqrtf(var + 1e-5f);
#pragma unroll
    for (int j = 0; j < 8; ++j) {
        int d = threadIdx.x + j * 256;
        float hn = (vals[j] - mean) * rstd * gamma[d] + beta[d];
        hnorm_b[(size_t)n * D_ + d] = f2bf(hn);
        out[(size_t)n * D_ + d]     = vals[j] + hn;
    }
}

// ------------------------------------------------------- leaky cascade scan
__global__ void scan_kernel(const float* __restrict__ alpha,
                            const float* __restrict__ xs,
                            unsigned short* __restrict__ casc) {
    int tid = blockIdx.x * 256 + threadIdx.x;   // 0..NTOK-1 state lanes
    int b  = tid >> 11;
    int r  = tid & 2047;
    int m  = r >> 8;
    int dc = r & (DC_ - 1);
    float s = 0.f;
    size_t base = (size_t)b * T_;
    for (int t = 0; t < T_; ++t) {
        size_t n = base + t;
        float a = alpha[n * M_ + m];
        float x = xs[n * DC_ + dc];
        s = a * s + (1.0f - a) * x;
        casc[n * 2048 + r] = f2bf(s);
    }
}

// ------------------------------------------------------------- WMMA GEMM
// C = A[N x Kd] @ Wt^T  with Wt stored K-major: Wt[p][k] (bf16).
// Tile: 64 rows x 256 cols per workgroup (8 wave32s, 2x4 WMMA per wave).
// EPI 0: outB = bf16(GELU(v + bias))
// EPI 1: outF = v (+bias if non-null)
// EPI 2: outF += bf2f(aux) * (v + bias)      (FiLM gain half)
// EPI 3: outF += v + bias                    (FiLM bias half)
// EPI 4: outF += v                           (cascade accumulation)
template <int EPI>
__global__ __launch_bounds__(256)
void gemm_bf16(const unsigned short* __restrict__ A, int Kd,
               const unsigned short* __restrict__ Wt, int wcolOff,
               const float* __restrict__ bias,
               const unsigned short* __restrict__ aux, int ldaux,
               float* __restrict__ outF, unsigned short* __restrict__ outB,
               int ldo) {
    // LDS row stride 80B (16B aligned, bank-spread). Double buffered.
    __shared__ unsigned int AL[2][64 * 20];    // [row][k-pair] uints
    __shared__ unsigned int BL[2][256 * 20];   // [col][k-pair] uints
    const int tid  = threadIdx.x;
    const int lane = tid & 31;
    const int wave = tid >> 5;
    const int rowTile = blockIdx.y * 64;
    const int colTile = blockIdx.x * 256;
    const int wr = (wave >> 2) * 32;           // 0 / 32
    const int wc = (wave & 3)  * 64;           // 0 / 64 / 128 / 192

    // staging: one 16B A chunk + four 16B B chunks per thread per slab
    const int ar = tid >> 2, ach = tid & 3;    // A row, chunk
    const int bc = tid >> 2, bch = tid & 3;    // B col base, chunk
    const char* Abase = (const char*)A + (size_t)(rowTile + ar) * Kd * 2 + ach * 16;
    const char* Bbase[4];
#pragma unroll
    for (int q = 0; q < 4; ++q)
        Bbase[q] = (const char*)Wt +
                   (size_t)(wcolOff + colTile + bc + q * 64) * Kd * 2 + bch * 16;

#if HAS_ASYNC
    auto issue = [&](int b, int k0) {
        async_copy16(Abase + k0 * 2, (char*)&AL[b][0] + ar * 80 + ach * 16);
#pragma unroll
        for (int q = 0; q < 4; ++q)
            async_copy16(Bbase[q] + k0 * 2,
                         (char*)&BL[b][0] + (bc + q * 64) * 80 + bch * 16);
    };
#else
    v4u ra, rb[4];
    auto loadRegs = [&](int k0) {
        ra = *(const v4u*)(Abase + k0 * 2);
#pragma unroll
        for (int q = 0; q < 4; ++q) rb[q] = *(const v4u*)(Bbase[q] + k0 * 2);
    };
    auto storeRegs = [&](int b) {
        *(v4u*)((char*)&AL[b][0] + ar * 80 + ach * 16) = ra;
#pragma unroll
        for (int q = 0; q < 4; ++q)
            *(v4u*)((char*)&BL[b][0] + (bc + q * 64) * 80 + bch * 16) = rb[q];
    };
#endif

    v8f acc[2][4] = {};
    auto compute = [&](int b) {
        const unsigned int* Au = &AL[b][0];
        const unsigned int* Bu = &BL[b][0];
        union F { v16bf v; unsigned int u[8]; };
        F af[2], bfr[4];
#pragma unroll
        for (int i = 0; i < 2; ++i) {          // A frags per ISA 16-bit A layout
            int r = wr + i * 16 + (lane & 15);
            int o = r * 20 + ((lane < 16) ? 0 : 4);
#pragma unroll
            for (int j = 0; j < 4; ++j) {
                af[i].u[j]     = Au[o + j];
                af[i].u[4 + j] = Au[o + 8 + j];
            }
        }
#pragma unroll
        for (int j = 0; j < 4; ++j) {          // B frags per ISA 16-bit B layout
            int c = wc + j * 16 + (lane & 15);
            int o = c * 20 + ((lane < 16) ? 0 : 4);
#pragma unroll
            for (int t = 0; t < 8; ++t) bfr[j].u[t] = Bu[o + t + (t & 4)];
        }
#pragma unroll
        for (int i = 0; i < 2; ++i)
#pragma unroll
            for (int j = 0; j < 4; ++j)
                acc[i][j] = __builtin_amdgcn_wmma_f32_16x16x32_bf16(
                    false, af[i].v, false, bfr[j].v, (short)0, acc[i][j], false, false);
    };

    const int nk = Kd >> 5;                    // >= 4 for all our shapes
#if HAS_ASYNC
    issue(0, 0);
#else
    loadRegs(0);
#endif
    for (int k = 0; k < nk - 1; ++k) {         // steady state: branch-free body
        const int b = k & 1;
#if HAS_ASYNC
        issue(b ^ 1, (k + 1) << 5);
        asm volatile("s_wait_asynccnt 0x5" ::: "memory");  // current slab done
#else
        storeRegs(b);
        loadRegs((k + 1) << 5);
#endif
        __syncthreads();
        compute(b);
        __syncthreads();
    }
#if HAS_ASYNC
    asm volatile("s_wait_asynccnt 0x0" ::: "memory");
#else
    storeRegs((nk - 1) & 1);
#endif
    __syncthreads();
    compute((nk - 1) & 1);

    // epilogue: lane<16 -> M=v ; lane>=16 -> M=8+v (per ISA C/D layout)
#pragma unroll
    for (int i = 0; i < 2; ++i) {
        int rbase = rowTile + wr + i * 16 + ((lane < 16) ? 0 : 8);
#pragma unroll
        for (int j = 0; j < 4; ++j) {
            int c = colTile + wc + j * 16 + (lane & 15);
            float bv = 0.0f;
            if (EPI == 0 || EPI == 2 || EPI == 3) bv = bias[wcolOff + c];
            else if (EPI == 1) { if (bias) bv = bias[wcolOff + c]; }
#pragma unroll
            for (int v = 0; v < 8; ++v) {
                int r = rbase + v;
                float val = acc[i][j][v] + bv;
                if (EPI == 0) {
                    float g = 0.5f * val * (1.0f + erff(val * 0.70710678f));
                    outB[(size_t)r * ldo + c] = f2bf(g);
                } else if (EPI == 1) {
                    outF[(size_t)r * ldo + c] = val;
                } else if (EPI == 2) {
                    float hn = bf2f(aux[(size_t)r * ldaux + c]);
                    outF[(size_t)r * ldo + c] += hn * val;
                } else if (EPI == 3) {
                    outF[(size_t)r * ldo + c] += val;
                } else {
                    outF[(size_t)r * ldo + c] += val;
                }
            }
        }
    }
}

// ---------------------------------------------------------------- launcher
extern "C" void kernel_launch(void* const* d_in, const int* in_sizes, int n_in,
                              void* d_out, int out_size, void* d_ws, size_t ws_size,
                              hipStream_t stream) {
    (void)in_sizes; (void)n_in; (void)out_size; (void)ws_size;
    const float* h      = (const float*)d_in[0];
    const float* delta  = (const float*)d_in[1];
    const float* mu     = (const float*)d_in[2];
    const float* lsig   = (const float*)d_in[3];
    const float* W1     = (const float*)d_in[4];
    const float* b1     = (const float*)d_in[5];
    const float* W2     = (const float*)d_in[6];
    const float* b2     = (const float*)d_in[7];
    const float* Wc_in  = (const float*)d_in[8];
    const float* Wc_out = (const float*)d_in[9];
    const float* gamma  = (const float*)d_in[10];
    const float* beta   = (const float*)d_in[11];
    float* out = (float*)d_out;

    char* ws = (char*)d_ws;
    size_t o = 0;
    auto carve = [&](size_t bytes) {
        char* p = ws + o;
        o = (o + bytes + 255) & ~(size_t)255;
        return p;
    };
    unsigned short* W1t    = (unsigned short*)carve((size_t)K_ * DC_ * 2);      // [DC][K]
    unsigned short* W2t    = (unsigned short*)carve((size_t)DC_ * 2 * D_ * 2);  // [2D][DC]
    unsigned short* Wcint  = (unsigned short*)carve((size_t)D_ * DC_ * 2);      // [DC][D]
    unsigned short* Wcoutt = (unsigned short*)carve((size_t)2048 * D_ * 2);     // [D][2048]
    unsigned short* tun_b  = (unsigned short*)carve((size_t)NTOK * K_ * 2);
    float*          alpha  = (float*)         carve((size_t)NTOK * M_ * 4);
    unsigned short* hid_b  = (unsigned short*)carve((size_t)NTOK * DC_ * 2);
    unsigned short* hnorm_b= (unsigned short*)carve((size_t)NTOK * D_ * 2);
    float*          xs     = (float*)         carve((size_t)NTOK * DC_ * 4);
    unsigned short* casc_b = (unsigned short*)carve((size_t)NTOK * 2048 * 2);

    // 1) weights -> bf16, transposed to K-major [P][Kd]
    transpose_bf16_kernel<<<dim3(DC_ / 32, K_ / 32),     256, 0, stream>>>(W1,     W1t,    K_,   DC_);
    transpose_bf16_kernel<<<dim3(2 * D_ / 32, DC_ / 32), 256, 0, stream>>>(W2,     W2t,    DC_,  2 * D_);
    transpose_bf16_kernel<<<dim3(DC_ / 32, D_ / 32),     256, 0, stream>>>(Wc_in,  Wcint,  D_,   DC_);
    transpose_bf16_kernel<<<dim3(D_ / 32, 2048 / 32),    256, 0, stream>>>(Wc_out, Wcoutt, 2048, D_);
    // 2) features
    tun_kernel  <<<NTOK * K_ / 256, 256, 0, stream>>>(delta, mu, lsig, tun_b);
    alpha_kernel<<<NTOK * M_ / 256, 256, 0, stream>>>(delta, alpha);
    // 3) hid = GELU(tun @ W1 + b1)
    gemm_bf16<0><<<dim3(DC_ / 256, NTOK / 64), 256, 0, stream>>>(
        tun_b, K_, W1t, 0, b1, nullptr, 0, nullptr, hid_b, DC_);
    // 4) layernorm; out = h + h_norm
    ln_kernel<<<NTOK, 256, 0, stream>>>(h, gamma, beta, hnorm_b, out);
    // 5a) out += h_norm * (hid @ W2[:, :D] + b2[:D])
    gemm_bf16<2><<<dim3(D_ / 256, NTOK / 64), 256, 0, stream>>>(
        hid_b, DC_, W2t, 0, b2, hnorm_b, D_, out, nullptr, D_);
    // 5b) out += hid @ W2[:, D:] + b2[D:]
    gemm_bf16<3><<<dim3(D_ / 256, NTOK / 64), 256, 0, stream>>>(
        hid_b, DC_, W2t, D_, b2, nullptr, 0, out, nullptr, D_);
    // 6) x_small = h_norm @ Wc_in
    gemm_bf16<1><<<dim3(DC_ / 256, NTOK / 64), 256, 0, stream>>>(
        hnorm_b, D_, Wcint, 0, nullptr, nullptr, 0, xs, nullptr, DC_);
    // 7) leaky cascade scan -> casc (bf16)
    scan_kernel<<<NTOK / 256, 256, 0, stream>>>(alpha, xs, casc_b);
    // 8) out += casc @ Wc_out
    gemm_bf16<4><<<dim3(D_ / 256, NTOK / 64), 256, 0, stream>>>(
        casc_b, 2048, Wcoutt, 0, nullptr, nullptr, 0, out, nullptr, D_);
}
